// Hypergraph_encoder_11948599017972
// MI455X (gfx1250) — matrix operand order repaired and checked by reading, not compiled, not run on hardware
//
#include <hip/hip_runtime.h>
#include <math.h>

// ---------------- types / WMMA helpers ----------------
typedef __attribute__((ext_vector_type(16))) _Float16 v16h;
typedef __attribute__((ext_vector_type(8)))  _Float16 h8v;
typedef __attribute__((ext_vector_type(8)))  float    v8f;
typedef __attribute__((ext_vector_type(4)))  float    f4v;

union AF16 { v16h v; _Float16 h[16]; h8v q[2]; };
union AF32 { v8f  v; float    f[8];  };

__device__ __forceinline__ float leakyf(float x) { return x > 0.f ? x : 0.01f * x; }

#define WMMA_F16(a, b, c) \
  __builtin_amdgcn_wmma_f32_16x16x32_f16(false, (a), false, (b), (short)0, (c), false, false)

// A fragment 16x32: lanes 0-15 M=lane&15; elem e -> K = (e&7) + 8*(2*(e>>3) + half)
// => per lane two contiguous 8-half runs at K = half*8 and K = 16 + half*8.
__device__ __forceinline__ v16h ldA_h(const _Float16* base, int ld, int k0) {
  int lane = threadIdx.x & 31, half = lane >> 4;
  const _Float16* rp = base + (size_t)(lane & 15) * ld + k0 + half * 8;
  AF16 a;
  a.q[0] = *(const h8v*)rp;
  a.q[1] = *(const h8v*)(rp + 16);
  return a.v;
}
// Same A fragment from f32 source (4x float4 loads + cvt)
__device__ __forceinline__ v16h ldA_f(const float* base, int ld, int k0) {
  int lane = threadIdx.x & 31, half = lane >> 4;
  const float* rp = base + (size_t)(lane & 15) * ld + k0 + half * 8;
  f4v x0 = *(const f4v*)rp;
  f4v x1 = *(const f4v*)(rp + 4);
  f4v x2 = *(const f4v*)(rp + 16);
  f4v x3 = *(const f4v*)(rp + 20);
  AF16 a;
#pragma unroll
  for (int j = 0; j < 4; ++j) {
    a.h[j]      = (_Float16)x0[j];
    a.h[4 + j]  = (_Float16)x1[j];
    a.h[8 + j]  = (_Float16)x2[j];
    a.h[12 + j] = (_Float16)x3[j];
  }
  return a.v;
}
// B fragment 32x16: N = lane&15, K = (e&7) + 8*(2*(e>>3) + half). Source stored K-contiguous per column
// (i.e. src[n*ld + k]); per lane two 8-half runs at K = half*8 and K = 16 + half*8 = two b128 loads.
__device__ __forceinline__ v16h ldB_h(const _Float16* base, int ld, int k0) {
  int lane = threadIdx.x & 31, half = lane >> 4;
  const _Float16* cp = base + (size_t)(lane & 15) * ld + k0 + 8 * half;
  AF16 b;
  b.q[0] = *(const h8v*)cp;
  b.q[1] = *(const h8v*)(cp + 16);
  return b.v;
}

// ---------------- constants ----------------
// B=8 T=12 N=512 D=64 E=ES=16 HS=16 HT=24 HTT=16
#define BT 96
#define NN 512
#define DD 64
#define XSZ 3145728     // B*T*N*D
#define HN  786432      // B*T*16*N

// ---------------- time feature MLPs ----------------
struct TFParams {
  const float *dW, *db, *wW, *wb, *l1W, *l1b, *l2W, *l2b, *loW, *lob;
};

__device__ void mlp16(const float* x, const TFParams p, float* out) {
  float h1[16], h2[16];
#pragma unroll
  for (int j = 0; j < 16; ++j) {
    float a = p.l1b[j];
    for (int i = 0; i < 16; ++i) a += x[i] * p.l1W[i * 16 + j];
    h1[j] = fmaxf(a, 0.f);
  }
#pragma unroll
  for (int j = 0; j < 16; ++j) {
    float a = p.l2b[j];
    for (int i = 0; i < 16; ++i) a += h1[i] * p.l2W[i * 16 + j];
    h2[j] = fmaxf(a, 0.f);
  }
#pragma unroll
  for (int j = 0; j < 16; ++j) {
    float a = p.lob[j];
    for (int i = 0; i < 16; ++i) a += h2[i] * p.loW[i * 16 + j];
    out[j] = a;
  }
}

__global__ void k_time(const float* __restrict__ source, TFParams tf1, TFParams tf1s,
                       TFParams tf2, float* time_eb, float* teb, float* te_spg) {
  int tid = threadIdx.x;
  if (tid < BT) {
    float day  = source[(size_t)tid * NN * 3 + 1];
    float week = source[(size_t)tid * NN * 3 + 2];
    float x[16];
#pragma unroll
    for (int j = 0; j < 16; ++j)
      x[j] = day * tf1.dW[j] + tf1.db[j] + week * tf1.wW[j] + tf1.wb[j];
    mlp16(x, tf1, time_eb + tid * 16);
#pragma unroll
    for (int j = 0; j < 16; ++j)
      x[j] = day * tf1s.dW[j] + tf1s.db[j] + week * tf1s.wW[j] + tf1s.wb[j];
    mlp16(x, tf1s, teb + tid * 16);
  } else if (tid < BT + 8) {
    int b = tid - BT;
    float x[16];
#pragma unroll
    for (int j = 0; j < 16; ++j) {
      float a = tf2.db[j] + tf2.wb[j];
      for (int t = 0; t < 12; ++t) {
        float day  = source[((size_t)(b * 12 + t) * NN) * 3 + 1];
        float week = source[((size_t)(b * 12 + t) * NN) * 3 + 2];
        a += day * tf2.dW[t * 16 + j] + week * tf2.wW[t * 16 + j];
      }
      x[j] = a;
    }
    mlp16(x, tf2, te_spg + b * 16);
  }
}

// ---------------- hyper_tem: rr = (adjT adj) eb, fused per node; emits f16 ----------------
__global__ void k_rr(const float* __restrict__ ne, const float* __restrict__ adj,
                     const float* __restrict__ x, _Float16* __restrict__ rrh) {
  __shared__ float s_ad[16 * 12];
  __shared__ float s_M[12 * 12];
  __shared__ float s_x[BT * DD];
  int n = blockIdx.x;
  for (int i = threadIdx.x; i < 192; i += 256) {
    int h = i / 12, t = i % 12;
    float a = 0.f;
    for (int k = 0; k < 16; ++k) a += ne[n * 16 + k] * adj[(k * 16 + h) * 12 + t];
    s_ad[i] = a;
  }
  for (int i = threadIdx.x; i < BT * DD; i += 256)
    s_x[i] = x[((size_t)(i >> 6) * NN + n) * DD + (i & 63)];
  __syncthreads();
  for (int i = threadIdx.x; i < 144; i += 256) {
    int t = i / 12, tp = i % 12;
    float m = 0.f;
    for (int h = 0; h < 16; ++h) m += s_ad[h * 12 + t] * s_ad[h * 12 + tp];
    s_M[i] = m;
  }
  __syncthreads();
  for (int i = threadIdx.x; i < BT * DD; i += 256) {
    int d = i & 63, bt = i >> 6, b = bt / 12, t = bt % 12;
    float acc = 0.f;
    for (int tp = 0; tp < 12; ++tp) acc += s_M[t * 12 + tp] * s_x[(b * 12 + tp) * DD + d];
    rrh[((size_t)bt * NN + n) * DD + d] = (_Float16)acc;
  }
}

// wT[row,o,i] = sum_d rv[row,d] * Wp[d,i,o]  (f16, pre-transposed for B fragments)
// bias[row,o] = sum_d rv[row,d] * bp[d,o]
__global__ void k_w(const float* __restrict__ rv, const float* __restrict__ Wp,
                    const float* __restrict__ bp, _Float16* __restrict__ wt,
                    float* __restrict__ bias) {
  int row = blockIdx.x;
  __shared__ float s[16];
  if (threadIdx.x < 16) s[threadIdx.x] = rv[row * 16 + threadIdx.x];
  __syncthreads();
  for (int j = threadIdx.x; j < 4096; j += 256) {
    int o = j >> 6, i = j & 63;
    float a = 0.f;
    for (int d = 0; d < 16; ++d) a += s[d] * Wp[d * 4096 + i * 64 + o];
    wt[(size_t)row * 4096 + j] = (_Float16)a;
  }
  if (threadIdx.x < 64) {
    float a = 0.f;
    for (int d = 0; d < 16; ++d) a += s[d] * bp[d * 64 + threadIdx.x];
    bias[row * 64 + threadIdx.x] = a;
  }
}

// wpt[o,i] = Wp[i,o] as f16 (64x64)
__global__ void k_prep_wpt(const float* __restrict__ Wp, _Float16* __restrict__ wpt) {
  int j = blockIdx.x * 256 + threadIdx.x;   // < 4096
  wpt[j] = (_Float16)Wp[(j & 63) * 64 + (j >> 6)];
}

// out[bt,n,o] = leaky( rr[bt]@w[bt] + bias + x ); per-bt 512x64 @ 64x64, f16 operands
__global__ void k_gemm_hyper(const _Float16* __restrict__ rrh, const _Float16* __restrict__ wt,
                             const float* __restrict__ bias, const float* __restrict__ x,
                             float* __restrict__ out) {
  int wid = (blockIdx.x * blockDim.x + threadIdx.x) >> 5;
  int nt = wid & 3, mt = (wid >> 2) & 31, bt = wid >> 7;
  const _Float16* A = rrh + (size_t)bt * NN * DD + (size_t)mt * 16 * DD;
  const _Float16* Bm = wt + (size_t)bt * 4096 + (size_t)(nt * 16) * 64;
  v8f acc = {};
#pragma unroll
  for (int k0 = 0; k0 < 64; k0 += 32)
    acc = WMMA_F16(ldA_h(A, DD, k0), ldB_h(Bm, DD, k0), acc);
  int lane = threadIdx.x & 31, half = lane >> 4, gcol = nt * 16 + (lane & 15);
  float bi = bias[bt * 64 + gcol];
  size_t obase = (size_t)bt * NN * DD + (size_t)mt * 16 * DD;
  AF32 u; u.v = acc;
#pragma unroll
  for (int r = 0; r < 8; ++r) {
    size_t idx = obase + (size_t)(r + 8 * half) * DD + gcol;
    out[idx] = leakyf(u.f[r] + bi + x[idx]);
  }
}

// Ppre = x @ Wp + bp ; 49152x64 @ 64x64, A from f32, B = wpt f16
__global__ void k_gemm_P(const float* __restrict__ x, const _Float16* __restrict__ wpt,
                         const float* __restrict__ bp, float* __restrict__ P) {
  int wid = (blockIdx.x * blockDim.x + threadIdx.x) >> 5;
  int nt = wid & 3, mt = wid >> 2;
  const float* A = x + (size_t)mt * 16 * DD;
  const _Float16* Bm = wpt + (size_t)(nt * 16) * 64;
  v8f acc = {};
#pragma unroll
  for (int k0 = 0; k0 < 64; k0 += 32)
    acc = WMMA_F16(ldA_f(A, DD, k0), ldB_h(Bm, DD, k0), acc);
  int lane = threadIdx.x & 31, half = lane >> 4, gcol = nt * 16 + (lane & 15);
  float bi = bp[gcol];
  AF32 u; u.v = acc;
#pragma unroll
  for (int r = 0; r < 8; ++r)
    P[(size_t)(mt * 16 + r + 8 * half) * DD + gcol] = u.f[r] + bi;
}

// squash(Ppre) -> Ph (n,d f16) and PhT (per bt: d-major, n contiguous? no: PhT[d][n], K-contig for B)
__global__ void k_squash_P(const float* __restrict__ Ppre, _Float16* __restrict__ ph,
                           _Float16* __restrict__ pht) {
  int r = blockIdx.x * blockDim.x + threadIdx.x;   // < BT*NN
  if (r >= BT * NN) return;
  int bt = r >> 9, n = r & 511;
  const float* pa = Ppre + (size_t)r * 64;
  float sn = 0.f;
  for (int d = 0; d < 64; ++d) { float v = pa[d]; sn += v * v; }
  float scale = sn / (1.f + sn) / (sqrtf(sn) + 1e-8f);
  for (int d = 0; d < 64; ++d) {
    float v = pa[d] * scale;
    ph[(size_t)r * 64 + d] = (_Float16)v;
    pht[(size_t)bt * 32768 + (size_t)d * 512 + n] = (_Float16)v;
  }
}

// generic f32 squash rows of 64; mode 0: x=a, 1: x=a*b, 2: x=a+b
__global__ void k_squash(const float* __restrict__ a, const float* __restrict__ b,
                         float* __restrict__ out, int rows, int mode) {
  int r = blockIdx.x * blockDim.x + threadIdx.x;
  if (r >= rows) return;
  const float* pa = a + (size_t)r * 64;
  const float* pb = b + (size_t)r * 64;
  float sn = 0.f;
  for (int d = 0; d < 64; ++d) {
    float v = pa[d];
    if (mode == 1) v *= pb[d]; else if (mode == 2) v += pb[d];
    sn += v * v;
  }
  float scale = sn / (1.f + sn) / (sqrtf(sn) + 1e-8f);
  float* po = out + (size_t)r * 64;
  for (int d = 0; d < 64; ++d) {
    float v = pa[d];
    if (mode == 1) v *= pb[d]; else if (mode == 2) v += pb[d];
    po[d] = v * scale;
  }
}

// v = squash(sq1 * cp) -> f16 row-major (A operand of uv GEMM)
__global__ void k_squash_VV(const float* __restrict__ sq1, const float* __restrict__ cp,
                            _Float16* __restrict__ vvh) {
  int r = blockIdx.x * blockDim.x + threadIdx.x;   // < BT*16
  if (r >= BT * 16) return;
  const float* pa = sq1 + (size_t)r * 64;
  const float* pb = cp + (size_t)r * 64;
  float sn = 0.f;
  for (int d = 0; d < 64; ++d) { float v = pa[d] * pb[d]; sn += v * v; }
  float scale = sn / (1.f + sn) / (sqrtf(sn) + 1e-8f);
  for (int d = 0; d < 64; ++d) vvh[(size_t)r * 64 + d] = (_Float16)(pa[d] * pb[d] * scale);
}

// v2 = squash(re + s) -> v2t f16 (bt, d, h) zero-padded to K=32 (B operand of recon GEMM)
__global__ void k_squash_V2(const float* __restrict__ re, const float* __restrict__ s,
                            _Float16* __restrict__ v2t) {
  int r = blockIdx.x * blockDim.x + threadIdx.x;   // < BT*16
  if (r >= BT * 16) return;
  int bt = r >> 4, h = r & 15;
  const float* pa = re + (size_t)r * 64;
  const float* pb = s + (size_t)r * 64;
  float sn = 0.f;
  for (int d = 0; d < 64; ++d) { float v = pa[d] + pb[d]; sn += v * v; }
  float scale = sn / (1.f + sn) / (sqrtf(sn) + 1e-8f);
  _Float16* base = v2t + (size_t)bt * 2048;
  for (int d = 0; d < 64; ++d) {
    base[d * 32 + h]      = (_Float16)((pa[d] + pb[d]) * scale);
    base[d * 32 + 16 + h] = (_Float16)0.f;
  }
}

// dadj[bt,h,n] = sum_d teb[bt,d] * adj[d,h,n]
__global__ void k_dadj(const float* __restrict__ teb, const float* __restrict__ adj,
                       float* __restrict__ dadj) {
  int i = blockIdx.x * 256 + threadIdx.x;
  int n = i & 511, h = (i >> 9) & 15, bt = i >> 13;
  float acc = 0.f;
  for (int d = 0; d < 16; ++d) acc += teb[bt * 16 + d] * adj[d * 8192 + h * 512 + n];
  dadj[i] = acc;
}

// softmax over h (stride 512) of f32 input -> f16 (h,n) layout (A operand)
__global__ void k_softmax_h(const float* __restrict__ in, _Float16* __restrict__ out, int count) {
  int i = blockIdx.x * blockDim.x + threadIdx.x;
  if (i >= count) return;
  int n = i & 511, bt = i >> 9;
  const float* p = in + (size_t)bt * 8192 + n;
  float mx = -1e30f;
  for (int h = 0; h < 16; ++h) mx = fmaxf(mx, p[h * 512]);
  float e[16], s = 0.f;
  for (int h = 0; h < 16; ++h) { e[h] = __expf(p[h * 512] - mx); s += e[h]; }
  float inv = 1.f / s;
  _Float16* o = out + (size_t)bt * 8192 + n;
  for (int h = 0; h < 16; ++h) o[h * 512] = (_Float16)(e[h] * inv);
}

// final c = softmax(bb + dadj): f16 (h,n), f16 transposed padded (n,32), and f32 to d_out
__global__ void k_softmax_h_add(const float* __restrict__ bb, const float* __restrict__ dadj,
                                _Float16* __restrict__ ch, _Float16* __restrict__ cth,
                                float* __restrict__ out2, int count) {
  int i = blockIdx.x * blockDim.x + threadIdx.x;
  if (i >= count) return;
  int n = i & 511, bt = i >> 9;
  size_t base = (size_t)bt * 8192 + n;
  float mx = -1e30f;
  for (int h = 0; h < 16; ++h) mx = fmaxf(mx, bb[base + h * 512] + dadj[base + h * 512]);
  float e[16], s = 0.f;
  for (int h = 0; h < 16; ++h) { e[h] = __expf(bb[base + h * 512] + dadj[base + h * 512] - mx); s += e[h]; }
  float inv = 1.f / s;
  _Float16* tp = cth + (size_t)bt * 16384 + (size_t)n * 32;
  for (int h = 0; h < 16; ++h) {
    float c = e[h] * inv;
    ch[base + h * 512] = (_Float16)c;
    tp[h] = (_Float16)c;
    tp[16 + h] = (_Float16)0.f;
    out2[base + h * 512] = c;
  }
}

// out[bt,h,d] = sum_n A[bt,h,n] * P[bt,n,d];  (16x512)@(512x64), B from PhT (d-major)
__global__ void k_gemm_hn(const _Float16* __restrict__ Amat, const _Float16* __restrict__ pht,
                          float* __restrict__ out) {
  int wid = (blockIdx.x * blockDim.x + threadIdx.x) >> 5;
  int nt = wid & 3, bt = wid >> 2;
  int lane = threadIdx.x & 31, l15 = lane & 15, half = lane >> 4;
  const _Float16* A = Amat + (size_t)bt * 8192;
  const _Float16* Bm = pht + (size_t)bt * 32768 + (size_t)(nt * 16) * 512;
  v8f acc = {};
  for (int k0 = 0; k0 < 512; k0 += 32) {
    if (k0 + 96 < 512) {
      __builtin_prefetch(A + (size_t)l15 * 512 + k0 + 96, 0, 1);
      __builtin_prefetch(Bm + (size_t)l15 * 512 + k0 + 96, 0, 1);
    }
    acc = WMMA_F16(ldA_h(A, 512, k0), ldB_h(Bm, 512, k0), acc);
  }
  int gcol = nt * 16 + l15;
  AF32 u; u.v = acc;
#pragma unroll
  for (int r = 0; r < 8; ++r)
    out[(size_t)bt * 1024 + (size_t)(r + 8 * half) * DD + gcol] = u.f[r];
}

// bb[bt,h,n] += sum_d V[bt,h,d] * P[bt,n,d];  B = Ph (n-major, d contiguous)
__global__ void k_gemm_uv(const _Float16* __restrict__ vvh, const _Float16* __restrict__ ph,
                          float* __restrict__ bb) {
  int wid = (blockIdx.x * blockDim.x + threadIdx.x) >> 5;
  int nt = wid & 31, bt = wid >> 5;
  int lane = threadIdx.x & 31, half = lane >> 4, col = lane & 15;
  const _Float16* A = vvh + (size_t)bt * 1024;
  const _Float16* Bm = ph + (size_t)bt * NN * DD + (size_t)(nt * 16) * 64;
  size_t bbase = (size_t)bt * 8192 + nt * 16;
  AF32 u;
#pragma unroll
  for (int r = 0; r < 8; ++r) u.f[r] = bb[bbase + (size_t)(r + 8 * half) * 512 + col];
  v8f acc = u.v;
#pragma unroll
  for (int k0 = 0; k0 < 64; k0 += 32)
    acc = WMMA_F16(ldA_h(A, DD, k0), ldB_h(Bm, DD, k0), acc);
  u.v = acc;
#pragma unroll
  for (int r = 0; r < 8; ++r) bb[bbase + (size_t)(r + 8 * half) * 512 + col] = u.f[r];
}

// recon[bt,n,d] = sum_h c[bt,h,n] * V2[bt,h,d];  A = cth (n,32 padded), B = v2t (d,32 padded)
__global__ void k_gemm_recon(const _Float16* __restrict__ cth, const _Float16* __restrict__ v2t,
                             _Float16* __restrict__ reconh) {
  int wid = (blockIdx.x * blockDim.x + threadIdx.x) >> 5;
  int nt = wid & 3, mt = (wid >> 2) & 31, bt = wid >> 7;
  const _Float16* A = cth + (size_t)bt * 16384 + (size_t)(mt * 16) * 32;
  const _Float16* Bm = v2t + (size_t)bt * 2048 + (size_t)(nt * 16) * 32;
  v8f acc = {};
  acc = WMMA_F16(ldA_h(A, 32, 0), ldB_h(Bm, 32, 0), acc);
  int lane = threadIdx.x & 31, half = lane >> 4, gcol = nt * 16 + (lane & 15);
  AF32 u; u.v = acc;
#pragma unroll
  for (int r = 0; r < 8; ++r)
    reconh[(size_t)bt * 32768 + (size_t)(mt * 16 + r + 8 * half) * 64 + gcol] = (_Float16)u.f[r];
}

// out[bt,n,o] = leaky( recon[:,n,:]@w_sp[n] + b_sp + x ); per-n 96x64 @ 64x64
__global__ void k_gemm_out(const _Float16* __restrict__ reconh, const _Float16* __restrict__ wspt,
                           const float* __restrict__ bsp, const float* __restrict__ x,
                           float* __restrict__ out) {
  int wid = (blockIdx.x * blockDim.x + threadIdx.x) >> 5;
  int nt = wid & 3, rest = wid >> 2, mt = rest % 6, n = rest / 6;
  const _Float16* A = reconh + (size_t)(mt * 16) * 32768 + (size_t)n * 64;  // row stride 512*64
  const _Float16* Bm = wspt + (size_t)n * 4096 + (size_t)(nt * 16) * 64;
  v8f acc = {};
#pragma unroll
  for (int k0 = 0; k0 < 64; k0 += 32)
    acc = WMMA_F16(ldA_h(A, 32768, k0), ldB_h(Bm, DD, k0), acc);
  int lane = threadIdx.x & 31, half = lane >> 4, gcol = nt * 16 + (lane & 15);
  float bi = bsp[n * 64 + gcol];
  AF32 u; u.v = acc;
#pragma unroll
  for (int r = 0; r < 8; ++r) {
    int btrow = mt * 16 + r + 8 * half;
    size_t idx = ((size_t)btrow * NN + n) * 64 + gcol;
    out[idx] = leakyf(u.f[r] + bi + x[idx]);
  }
}

// ---------------- small helpers ----------------
__global__ void k_zero(float* p, int count) {
  int i = blockIdx.x * blockDim.x + threadIdx.x;
  if (i < count) p[i] = 0.f;
}
__global__ void k_hs(const float* __restrict__ s, float* __restrict__ hs) {
  int i = blockIdx.x * 256 + threadIdx.x;          // < 98304
  int t = (i >> 10) % 12;
  hs[i] = s[i] + (float)(t + 1) / 12.0f;
}
__global__ void k_dyn(const float* __restrict__ te_spg, const float* __restrict__ t_adj,
                      float* __restrict__ dyn) {
  int i = blockIdx.x * 256 + threadIdx.x;          // < 8*24*192
  int k = i % 192, h = (i / 192) % 24, b = i / 4608;
  float acc = 0.f;
  for (int d = 0; d < 16; ++d) acc += te_spg[b * 16 + d] * t_adj[d * 4608 + h * 192 + k];
  dyn[i] = acc;
}
__global__ void k_he(const float* __restrict__ dyn, const float* __restrict__ hs,
                     float* __restrict__ he) {
  int i = blockIdx.x * 256 + threadIdx.x;          // < 8*24*64
  int d = i & 63, h = (i >> 6) % 24, b = i / 1536;
  float acc = 0.f;
  for (int k = 0; k < 192; ++k) acc += dyn[b * 4608 + h * 192 + k] * hs[(size_t)b * 12288 + k * 64 + d];
  he[i] = leakyf(acc);
}
__global__ void k_re(const float* __restrict__ dyn, const float* __restrict__ he,
                     float* __restrict__ re) {
  int i = blockIdx.x * 256 + threadIdx.x;          // < 8*192*64
  int d = i & 63, k = (i >> 6) % 192, b = i / 12288;
  float acc = 0.f;
  for (int h = 0; h < 24; ++h) acc += dyn[b * 4608 + h * 192 + k] * he[b * 1536 + h * 64 + d];
  re[i] = leakyf(acc);
}

// ---------------- host ----------------
static TFParams mkTF(void* const* d_in, int base) {
  TFParams p;
  p.dW  = (const float*)d_in[base + 0]; p.db  = (const float*)d_in[base + 1];
  p.wW  = (const float*)d_in[base + 2]; p.wb  = (const float*)d_in[base + 3];
  p.l1W = (const float*)d_in[base + 4]; p.l1b = (const float*)d_in[base + 5];
  p.l2W = (const float*)d_in[base + 6]; p.l2b = (const float*)d_in[base + 7];
  p.loW = (const float*)d_in[base + 8]; p.lob = (const float*)d_in[base + 9];
  return p;
}

extern "C" void kernel_launch(void* const* d_in, const int* in_sizes, int n_in,
                              void* d_out, int out_size, void* d_ws, size_t ws_size,
                              hipStream_t stream) {
  (void)in_sizes; (void)n_in; (void)out_size; (void)ws_size;
  const float* source = (const float*)d_in[0];
  const float* x_in   = (const float*)d_in[1];
  const float* ne     = (const float*)d_in[2];
  const float* ne_spg = (const float*)d_in[3];
  TFParams tf1  = mkTF(d_in, 4);
  TFParams tf1s = mkTF(d_in, 14);
  TFParams tf2  = mkTF(d_in, 24);
  // ht i: base 34 + 3*i (adj, Wp, bp); cap1 base 46, cap2 base 52 (Wp, bp, t_adj, adj, Wspa, bspa)

  float* ws = (float*)d_ws;
  size_t off = 0;
  auto allocF = [&](size_t n) { float* p = ws + off; off += n; return p; };
  auto allocH = [&](size_t n) { _Float16* p = (_Float16*)(ws + off); off += n / 2; return p; };

  float*    TIMEEB = allocF(BT * 16);
  float*    TEB    = allocF(BT * 16);
  float*    TESPG  = allocF(128);
  _Float16* WTH    = allocH((size_t)NN * 4096);   // shared: per-bt wT (96x4096) / w_spT (512x4096)
  float*    BIASBT = allocF(NN * 64);
  _Float16* RRH    = allocH(XSZ);
  float*    XT     = allocF(XSZ);
  float*    G      = allocF(XSZ);
  float*    PPRE   = allocF(XSZ);
  _Float16* PH     = allocH(XSZ);
  _Float16* PHT    = allocH(XSZ);
  float*    DADJ   = allocF(HN);
  float*    BB     = allocF(HN);
  _Float16* CH     = allocH(HN);
  _Float16* CTH    = allocH((size_t)BT * 512 * 32);
  float*    SQ1    = allocF(BT * 1024);
  float*    CP     = allocF(BT * 1024);
  _Float16* VVH    = allocH(BT * 1024);
  float*    S      = allocF(BT * 1024);
  float*    HSB    = allocF(BT * 1024);
  float*    DYN    = allocF(8 * 24 * 192);
  float*    HE     = allocF(8 * 24 * 64);
  float*    RE     = allocF(8 * 192 * 64);
  _Float16* V2T    = allocH(BT * 2048);
  _Float16* RECONH = allocH(XSZ);
  _Float16* WPT    = allocH(4096);
  float*    BSP    = allocF(NN * 64);

  float* dout = (float*)d_out;
  float* HS1_out = dout + XSZ;
  float* HS3_out = dout + XSZ + HN;

  k_time<<<1, 128, 0, stream>>>(source, tf1, tf1s, tf2, TIMEEB, TEB, TESPG);

  auto hyper = [&](const float* xsrc, int base, float* dst) {
    const float* adj = (const float*)d_in[base + 0];
    const float* Wp  = (const float*)d_in[base + 1];
    const float* bp  = (const float*)d_in[base + 2];
    k_rr<<<NN, 256, 0, stream>>>(ne, adj, xsrc, RRH);
    k_w<<<BT, 256, 0, stream>>>(TIMEEB, Wp, bp, WTH, BIASBT);
    k_gemm_hyper<<<1536, 256, 0, stream>>>(RRH, WTH, BIASBT, xsrc, dst);
  };

  auto cap = [&](const float* xsrc, int base, float* dst, float* c_out) {
    const float* cWp  = (const float*)d_in[base + 0];
    const float* cbp  = (const float*)d_in[base + 1];
    const float* tadj = (const float*)d_in[base + 2];
    const float* cadj = (const float*)d_in[base + 3];
    const float* Wspa = (const float*)d_in[base + 4];
    const float* bspa = (const float*)d_in[base + 5];
    k_prep_wpt<<<16, 256, 0, stream>>>(cWp, WPT);
    k_gemm_P<<<1536, 256, 0, stream>>>(xsrc, WPT, cbp, PPRE);
    k_squash_P<<<192, 256, 0, stream>>>(PPRE, PH, PHT);                 // P = squash(xWp+bp)
    k_dadj<<<3072, 256, 0, stream>>>(TEB, cadj, DADJ);
    k_softmax_h<<<192, 256, 0, stream>>>(DADJ, CH, BT * NN);            // softmax(dadj)
    k_gemm_hn<<<48, 256, 0, stream>>>(CH, PHT, CP);                     // test1
    k_squash<<<6, 256, 0, stream>>>(CP, CP, SQ1, BT * 16, 0);           // sq1
    k_zero<<<3072, 256, 0, stream>>>(BB, HN);
    for (int it = 0; it < 3; ++it) {
      k_softmax_h<<<192, 256, 0, stream>>>(BB, CH, BT * NN);            // c
      k_gemm_hn<<<48, 256, 0, stream>>>(CH, PHT, CP);                   // c @ P
      k_squash_VV<<<6, 256, 0, stream>>>(SQ1, CP, VVH);                 // v = squash(sq1*cp)
      k_gemm_uv<<<384, 256, 0, stream>>>(VVH, PH, BB);                  // bb += v @ P^T
    }
    k_softmax_h_add<<<192, 256, 0, stream>>>(BB, DADJ, CH, CTH, c_out, BT * NN);
    k_gemm_hn<<<48, 256, 0, stream>>>(CH, PHT, S);                      // s = c @ P
    k_hs<<<384, 256, 0, stream>>>(S, HSB);
    k_dyn<<<144, 256, 0, stream>>>(TESPG, tadj, DYN);
    k_he<<<48, 256, 0, stream>>>(DYN, HSB, HE);
    k_re<<<384, 256, 0, stream>>>(DYN, HE, RE);
    k_squash_V2<<<6, 256, 0, stream>>>(RE, S, V2T);                     // v2 = squash(re+s), padded T
    k_gemm_recon<<<1536, 256, 0, stream>>>(CTH, V2T, RECONH);           // recon (f16)
    k_w<<<NN, 256, 0, stream>>>(ne_spg, Wspa, bspa, WTH, BSP);          // w_spT / b_sp
    k_gemm_out<<<1536, 256, 0, stream>>>(RECONH, WTH, BSP, xsrc, dst);
  };

  hyper(x_in, 34, XT);          // xt1
  cap(XT, 46, G, HS1_out);      // g1, HS1
  hyper(G, 37, XT);             // xt2
  hyper(XT, 40, G);             // xt3
  cap(G, 52, XT, HS3_out);      // g3, HS3
  hyper(XT, 43, dout);          // xt4 -> d_out
}
